// VarianceAdaptor_8031588843965
// MI455X (gfx1250) — matrix operand order repaired and checked by reading, not compiled, not run on hardware
//
#include <hip/hip_runtime.h>

typedef _Float16 h16;
typedef __attribute__((ext_vector_type(16))) _Float16 v16h;
typedef __attribute__((ext_vector_type(8)))  float    v8f;

#define HD   384
#define TSEQ 1024
#define BB   32
#define MAXL 4096
#define INSTR 392   // LDS input tile row stride (halfs), 784B: conflict-free b128 reads
#define OSTR  388   // LDS output tile row stride (halfs)

__device__ __forceinline__ float wsum(float v) {
#pragma unroll
  for (int m = 16; m >= 1; m >>= 1) v += __shfl_xor(v, m, 32);
  return v;
}

union AF { v16h v; float4 f[2]; };

// ---------------------------------------------------------------------------
// Fused conv1d(k) + bias + ReLU + LayerNorm(H) using v_wmma_f32_16x16x32_f16.
// One block: 32 time steps x 384 out-channels of one batch row.
// mode 0: f16 src buffer; mode 1: f32 src (x); mode 2: f32 src + alpha*PE(pos)
// ---------------------------------------------------------------------------
template <int KT>
__global__ __launch_bounds__(256) void conv_ln_kernel(
    const float* __restrict__ xsrc, const h16* __restrict__ hsrc,
    const h16* __restrict__ wpack, const float* __restrict__ bias,
    const float* __restrict__ gma, const float* __restrict__ bta,
    const int* __restrict__ pos, const float* __restrict__ alpha,
    h16* __restrict__ dst, int mode)
{
  constexpr int P = (KT - 1) / 2;
  constexpr int R = 32 + KT - 1;
  extern __shared__ char smem[];
  h16* sOut = (h16*)smem;                      // 32 x OSTR halfs
  h16* sIn  = (h16*)(smem + 32 * OSTR * 2);    // R  x INSTR halfs

  const int b  = blockIdx.x >> 5;
  const int t0 = (blockIdx.x & 31) << 5;

  // Stage input tile (with 'same' zero padding, optional PE) into LDS as f16.
  for (int idx = threadIdx.x; idx < R * HD; idx += 256) {
    int row = idx / HD;
    int ch  = idx - row * HD;
    int t   = t0 - P + row;
    float v = 0.f;
    if (t >= 0 && t < TSEQ) {
      int off = (b * TSEQ + t) * HD + ch;
      if (mode == 0) v = (float)hsrc[off];
      else {
        v = xsrc[off];
        if (mode == 2) {
          int pp = pos[b * TSEQ + t];
          if (pp != 0) {
            int i2 = (ch < 192) ? ch : (ch - 192);
            float ang = (float)pp * expf(-0.048221677f * (float)i2); // ln(1e4)/191
            v += alpha[0] * ((ch < 192) ? sinf(ang) : cosf(ang));
          }
        }
      }
    }
    sIn[row * INSTR + ch] = (h16)v;
  }
  __syncthreads();

  const int wave = threadIdx.x >> 5;
  const int lane = threadIdx.x & 31;
  const int lm   = lane & 15;
  const int hi8  = (lane >> 4) * 8;
  const int mtile = wave >> 2;   // 0..1: rows 0-15 / 16-31
  const int ng    = wave & 3;    // 6 consecutive N-tiles per wave

  v8f acc[6] = {};
  for (int tap = 0; tap < KT; ++tap) {
    for (int kch = 0; kch < 12; ++kch) {
      AF a;  // A 16x32 f16 fragment: lanes<16 K0..7/16..23, lanes>=16 K8..15/24..31
      const h16* ap = sIn + (mtile * 16 + lm + tap) * INSTR + kch * 32 + hi8;
      a.f[0] = *(const float4*)(ap);
      a.f[1] = *(const float4*)(ap + 16);
#pragma unroll
      for (int j = 0; j < 6; ++j) {
        int ntile = ng * 6 + j;
        const h16* bp = wpack + ((((tap * 12 + kch) * 24 + ntile) << 9) + lane * 8);
        AF bf;  // pre-swizzled: 2 contiguous 512B bursts per fragment
        bf.f[0] = *(const float4*)(bp);
        bf.f[1] = *(const float4*)(bp + 256);
        acc[j] = __builtin_amdgcn_wmma_f32_16x16x32_f16(
            false, a.v, false, bf.v, (short)0, acc[j], false, false);
      }
    }
  }

  // bias + ReLU, stage to LDS (row = time, col = channel) for LayerNorm
#pragma unroll
  for (int j = 0; j < 6; ++j) {
    int n = (ng * 6 + j) * 16 + lm;
    float bv = bias[n];
#pragma unroll
    for (int r = 0; r < 8; ++r) {
      int m = mtile * 16 + hi8 + r;
      float v = acc[j][r] + bv;
      sOut[m * OSTR + n] = (h16)(v > 0.f ? v : 0.f);
    }
  }
  __syncthreads();

  // LayerNorm over H per time step; 4 rows per wave, shuffle reduction.
  for (int rr = 0; rr < 4; ++rr) {
    int m = wave * 4 + rr;
    float vals[12], s = 0.f, s2 = 0.f;
#pragma unroll
    for (int i = 0; i < 12; ++i) {
      float v = (float)sOut[m * OSTR + lane + 32 * i];
      vals[i] = v; s += v; s2 += v * v;
    }
    s = wsum(s); s2 = wsum(s2);
    float mean = s * (1.f / 384.f);
    float var  = s2 * (1.f / 384.f) - mean * mean;
    float rstd = rsqrtf(var + 1e-5f);
    int rowoff = (b * TSEQ + t0 + m) * HD;
#pragma unroll
    for (int i = 0; i < 12; ++i) {
      int n = lane + 32 * i;
      dst[rowoff + n] = (h16)((vals[i] - mean) * rstd * gma[n] + bta[n]);
    }
  }
}

// Pack conv weight (O,I,k) f32 -> per-WMMA-fragment lane-major f16 layout.
__global__ void pack_w_kernel(const float* __restrict__ w, h16* __restrict__ dst,
                              int ktaps, int total)
{
  int gid = blockIdx.x * 256 + threadIdx.x;
  if (gid >= total) return;
  int h = gid & 7;
  int l = (gid >> 3) & 31;
  int p = (gid >> 8) & 1;
  int fid = gid >> 9;
  int ntile = fid % 24;
  int tmp   = fid / 24;
  int kch   = tmp % 12;
  int tap   = tmp / 12;
  int K = kch * 32 + p * 16 + ((l >= 16) ? 8 : 0) + h;  // B 32x16: lanes<16 K0..7/16..23
  int n = ntile * 16 + (l & 15);
  dst[gid] = (h16)w[(n * HD + K) * ktaps + tap];
}

// Linear head (H->1). mode 0: duration (mask, exp/round). mode 1: variance
// (searchsorted 255 bins + embedding add into x_adapt). One wave per position.
__global__ __launch_bounds__(256) void head_kernel(
    const h16* __restrict__ act, const float* __restrict__ lw,
    const float* __restrict__ lb, const unsigned char* __restrict__ mask,
    const float* __restrict__ bins, const float* __restrict__ table,
    float* __restrict__ pred, float* __restrict__ dur,
    float* __restrict__ xad, int mode)
{
  int wave = threadIdx.x >> 5, lane = threadIdx.x & 31;
  int p = blockIdx.x * 8 + wave;
  const h16* row = act + p * HD;
  float s = 0.f;
#pragma unroll
  for (int i = 0; i < 12; ++i) { int c = lane + 32 * i; s += (float)row[c] * lw[c]; }
  s = wsum(s);
  float v = s + lb[0];
  if (mode == 0) {
    if (mask[p]) v = 0.f;
    if (lane == 0) { pred[p] = v; dur[p] = fmaxf(0.f, roundf(expf(v) - 1.f)); }
  } else {
    if (lane == 0) pred[p] = v;
    int lo = 0, hiN = 255;                       // side='left'
    while (lo < hiN) { int mid = (lo + hiN) >> 1; if (bins[mid] < v) lo = mid + 1; else hiN = mid; }
    const float* tr = table + lo * HD;
    float* xr = xad + p * HD;
#pragma unroll
    for (int i = 0; i < 12; ++i) { int c = lane + 32 * i; xr[c] += tr[c]; }
  }
}

// pos = cumsum(x[...,0]!=0) * mask, per batch (Hillis-Steele scan over 1024)
__global__ void pos_kernel(const float* __restrict__ x, int* __restrict__ pos)
{
  __shared__ int s[1024];
  int b = blockIdx.x, t = threadIdx.x;
  int m = (x[(b * TSEQ + t) * HD] != 0.f) ? 1 : 0;
  s[t] = m;
  __syncthreads();
  for (int off = 1; off < 1024; off <<= 1) {
    int add = (t >= off) ? s[t - off] : 0;
    __syncthreads();
    s[t] += add;
    __syncthreads();
  }
  pos[b * TSEQ + t] = s[t] * m;
}

__global__ void cum_kernel(const float* __restrict__ dur, int* __restrict__ cum,
                           float* __restrict__ mel)
{
  __shared__ int s[1024];
  int b = blockIdx.x, t = threadIdx.x;
  s[t] = (int)dur[b * TSEQ + t];
  __syncthreads();
  for (int off = 1; off < 1024; off <<= 1) {
    int add = (t >= off) ? s[t - off] : 0;
    __syncthreads();
    s[t] += add;
    __syncthreads();
  }
  cum[b * TSEQ + t] = s[t];
  if (t == 1023) { int tot = s[t]; mel[b] = (float)(tot < MAXL ? tot : MAXL); }
}

// Length regulation: 64 frames per block, binary search into cached cumsum.
__global__ __launch_bounds__(256) void regulate_kernel(
    const float* __restrict__ xad, const int* __restrict__ cum,
    float* __restrict__ out)
{
  __shared__ int sc[1024];
  __shared__ int sidx[64];
  int b  = blockIdx.x >> 6;
  int f0 = (blockIdx.x & 63) << 6;
  for (int i = threadIdx.x; i < 1024; i += 256) sc[i] = cum[b * TSEQ + i];
  __syncthreads();
  if (threadIdx.x < 64) {
    int f = f0 + threadIdx.x;
    int lo = 0, hiN = 1024;                      // side='right'
    while (lo < hiN) { int mid = (lo + hiN) >> 1; if (sc[mid] <= f) lo = mid + 1; else hiN = mid; }
    int idx = lo < 1023 ? lo : 1023;
    sidx[threadIdx.x] = (f < sc[1023]) ? idx : -1;
  }
  __syncthreads();
  for (int e = threadIdx.x; e < 64 * HD; e += 256) {
    int fr = e / HD, ch = e - fr * HD;
    int ix = sidx[fr];
    out[(b * MAXL + f0 + fr) * HD + ch] = (ix >= 0) ? xad[(b * TSEQ + ix) * HD + ch] : 0.f;
  }
}

extern "C" void kernel_launch(void* const* d_in, const int* in_sizes, int n_in,
                              void* d_out, int out_size, void* d_ws, size_t ws_size,
                              hipStream_t stream)
{
  (void)in_sizes; (void)n_in; (void)out_size; (void)ws_size;
  const float* x        = (const float*)d_in[0];
  const unsigned char* src_mask = (const unsigned char*)d_in[2];
  const float* dp_w1 = (const float*)d_in[4];
  const float* dp_b1 = (const float*)d_in[5];
  const float* dp_g1 = (const float*)d_in[6];
  const float* dp_be1= (const float*)d_in[7];
  const float* dp_w2 = (const float*)d_in[8];
  const float* dp_b2 = (const float*)d_in[9];
  const float* dp_g2 = (const float*)d_in[10];
  const float* dp_be2= (const float*)d_in[11];
  const float* dp_lw = (const float*)d_in[12];
  const float* dp_lb = (const float*)d_in[13];
  const float* pp_cw = (const float*)d_in[14];
  const float* pp_cb = (const float*)d_in[15];
  const float* pp_g  = (const float*)d_in[16];
  const float* pp_b  = (const float*)d_in[17];
  const float* pp_lw = (const float*)d_in[18];
  const float* pp_lb = (const float*)d_in[19];
  const float* pp_alpha = (const float*)d_in[20];
  const float* ep_cw = (const float*)d_in[21];
  const float* ep_cb = (const float*)d_in[22];
  const float* ep_g  = (const float*)d_in[23];
  const float* ep_b  = (const float*)d_in[24];
  const float* ep_lw = (const float*)d_in[25];
  const float* ep_lb = (const float*)d_in[26];
  const float* ep_alpha = (const float*)d_in[27];
  const float* pitch_bins   = (const float*)d_in[28];
  const float* energy_bins  = (const float*)d_in[29];
  const float* pitch_table  = (const float*)d_in[30];
  const float* energy_table = (const float*)d_in[31];

  // workspace carve-up
  char* ws = (char*)d_ws;
  size_t off = 0;
  auto alloc = [&](size_t bytes) -> char* {
    char* r = ws + off; off += (bytes + 255) & ~(size_t)255; return r;
  };
  const size_t NPOS = (size_t)BB * TSEQ;          // 32768
  int*   pos  = (int*)alloc(NPOS * 4);
  int*   cum  = (int*)alloc(NPOS * 4);
  h16*   tmpA = (h16*)alloc(NPOS * HD * 2);
  h16*   tmpB = (h16*)alloc(NPOS * HD * 2);
  float* xad  = (float*)alloc(NPOS * HD * 4);
  const int PK3 = 24 * 3 * 12 * 512;              // 442368 halfs
  const int PK5 = 24 * 5 * 12 * 512;              // 737280 halfs
  h16* wdp1 = (h16*)alloc((size_t)PK3 * 2);
  h16* wdp2 = (h16*)alloc((size_t)PK3 * 2);
  h16* wpp[5];
  for (int L = 0; L < 5; ++L) wpp[L] = (h16*)alloc((size_t)PK5 * 2);
  h16* wep[2];
  for (int L = 0; L < 2; ++L) wep[L] = (h16*)alloc((size_t)PK5 * 2);

  // output slices (concatenated, return order)
  float* o        = (float*)d_out;
  float* O_out    = o;
  float* O_logd   = o + (size_t)BB * MAXL * HD;
  float* O_dur    = O_logd + NPOS;
  float* O_pitch  = O_dur + NPOS;
  float* O_energy = O_pitch + NPOS;
  float* O_mel    = O_energy + NPOS;

  // --- weight packing (cheap, deterministic every call) ---
  pack_w_kernel<<<(PK3 + 255) / 256, 256, 0, stream>>>(dp_w1, wdp1, 3, PK3);
  pack_w_kernel<<<(PK3 + 255) / 256, 256, 0, stream>>>(dp_w2, wdp2, 3, PK3);
  for (int L = 0; L < 5; ++L)
    pack_w_kernel<<<(PK5 + 255) / 256, 256, 0, stream>>>(pp_cw + (size_t)L * HD * HD * 5, wpp[L], 5, PK5);
  for (int L = 0; L < 2; ++L)
    pack_w_kernel<<<(PK5 + 255) / 256, 256, 0, stream>>>(ep_cw + (size_t)L * HD * HD * 5, wep[L], 5, PK5);

  pos_kernel<<<BB, 1024, 0, stream>>>(x, pos);
  hipMemcpyAsync(xad, x, NPOS * HD * 4, hipMemcpyDeviceToDevice, stream);

  const size_t sh3 = 32 * OSTR * 2 + (size_t)(32 + 2) * INSTR * 2;
  const size_t sh5 = 32 * OSTR * 2 + (size_t)(32 + 4) * INSTR * 2;

  // --- duration predictor ---
  conv_ln_kernel<3><<<1024, 256, sh3, stream>>>(x, nullptr, wdp1, dp_b1, dp_g1, dp_be1,
                                                nullptr, nullptr, tmpA, 1);
  conv_ln_kernel<3><<<1024, 256, sh3, stream>>>(nullptr, tmpA, wdp2, dp_b2, dp_g2, dp_be2,
                                                nullptr, nullptr, tmpB, 0);
  head_kernel<<<4096, 256, 0, stream>>>(tmpB, dp_lw, dp_lb, src_mask,
                                        nullptr, nullptr, O_logd, O_dur, nullptr, 0);

  // --- pitch predictor (5 layers, ping-pong) ---
  conv_ln_kernel<5><<<1024, 256, sh5, stream>>>(x, nullptr, wpp[0], pp_cb, pp_g, pp_b,
                                                pos, pp_alpha, tmpA, 2);
  conv_ln_kernel<5><<<1024, 256, sh5, stream>>>(nullptr, tmpA, wpp[1], pp_cb + HD, pp_g + HD, pp_b + HD,
                                                nullptr, nullptr, tmpB, 0);
  conv_ln_kernel<5><<<1024, 256, sh5, stream>>>(nullptr, tmpB, wpp[2], pp_cb + 2 * HD, pp_g + 2 * HD, pp_b + 2 * HD,
                                                nullptr, nullptr, tmpA, 0);
  conv_ln_kernel<5><<<1024, 256, sh5, stream>>>(nullptr, tmpA, wpp[3], pp_cb + 3 * HD, pp_g + 3 * HD, pp_b + 3 * HD,
                                                nullptr, nullptr, tmpB, 0);
  conv_ln_kernel<5><<<1024, 256, sh5, stream>>>(nullptr, tmpB, wpp[4], pp_cb + 4 * HD, pp_g + 4 * HD, pp_b + 4 * HD,
                                                nullptr, nullptr, tmpA, 0);
  head_kernel<<<4096, 256, 0, stream>>>(tmpA, pp_lw, pp_lb, nullptr,
                                        pitch_bins, pitch_table, O_pitch, nullptr, xad, 1);

  // --- energy predictor (2 layers) ---
  conv_ln_kernel<5><<<1024, 256, sh5, stream>>>(x, nullptr, wep[0], ep_cb, ep_g, ep_b,
                                                pos, ep_alpha, tmpB, 2);
  conv_ln_kernel<5><<<1024, 256, sh5, stream>>>(nullptr, tmpB, wep[1], ep_cb + HD, ep_g + HD, ep_b + HD,
                                                nullptr, nullptr, tmpA, 0);
  head_kernel<<<4096, 256, 0, stream>>>(tmpA, ep_lw, ep_lb, nullptr,
                                        energy_bins, energy_table, O_energy, nullptr, xad, 1);

  // --- length regulate ---
  cum_kernel<<<BB, 1024, 0, stream>>>(O_dur, cum, O_mel);
  regulate_kernel<<<BB * 64, 256, 0, stream>>>(xad, cum, O_out);
}
